// RNNCellBuilder_78795470012673
// MI455X (gfx1250) — matrix operand order repaired and checked by reading, not compiled, not run on hardware
//
#include <hip/hip_runtime.h>
#include <hip/hip_bf16.h>
#include <math.h>

// ---------------------------------------------------------------------------
// GRU cell on MI455X (gfx1250), bf16 WMMA with f32 accumulate.
//   z  = sigmoid(x@Wz_x + h@Wz_h + bz)            (GEMM mode 0 -> ws)
//   rh = sigmoid(x@Wr_x + h@Wr_h + br) * h        (GEMM mode 1 -> ws)
//   hn = z*h + (1-z)*tanh(x@Wh_x + rh@Wh_rh + bh) (GEMM mode 2 -> out)
// All three are dual-K GEMMs with K = 1024 + 1024, M=4096, N=1024.
//
// Block tile 128x128, staged slab BK=64 (2 WMMA K-steps per barrier pair).
// 8 waves of 32 lanes: 2(M) x 4(N) wave grid, each wave owns 64x32 =
// 4 M-frags x 2 N-frags -> 8 WMMAs per K-step, 16 per stage.
// ---------------------------------------------------------------------------

typedef __bf16 v16bf __attribute__((ext_vector_type(16)));
typedef float  v8f   __attribute__((ext_vector_type(8)));
typedef float  f4    __attribute__((ext_vector_type(4)));

#define NROWS 4096
#define DU    1024   // lda/ldb/ldc for every matrix here

__device__ __forceinline__ unsigned pack_bf16x2(float lo, float hi) {
    union { __bf16 h[2]; unsigned u; } r;
    r.h[0] = (__bf16)lo;
    r.h[1] = (__bf16)hi;
    return r.u;
}

template <int MODE>
__global__ __launch_bounds__(256)
void gru_gemm(const float* __restrict__ A1, const float* __restrict__ W1,
              const float* __restrict__ A2, const float* __restrict__ W2,
              const float* __restrict__ bias,
              const float* __restrict__ hbuf,   // MODE 1,2
              const float* __restrict__ zbuf,   // MODE 2
              float* __restrict__ out)
{
    // Fragment-swizzled LDS: slot = (kstep, frag, lane) -> 16 contiguous bf16
    // (32B) == one v16bf WMMA operand for that lane.
    __shared__ __align__(64) __bf16 ldsA[2 * 8 * 32 * 16]; // 16 KB
    __shared__ __align__(64) __bf16 ldsB[2 * 8 * 32 * 16]; // 16 KB

    const int t      = threadIdx.x;
    const int lane   = t & 31;
    const int wave   = t >> 5;
    const int waveM  = wave >> 2;  // 0..1 -> 64-row slice
    const int waveN  = wave & 3;   // 0..3 -> 32-col slice
    const int blockN = blockIdx.x; // 0..7   (cols,  1024/128)
    const int blockM = blockIdx.y; // 0..31  (rows,  4096/128)

    // --- A staging: thread t owns fragment slot (t>>5, t&31) in both ksteps.
    // ISA 16-bit A 16x32 layout: lane<16 holds K {0..7,16..23}; lane>=16 {8..15,24..31}.
    const int afrag = t >> 5;
    const int am    = (afrag << 4) + (lane & 15);
    const int ab0   = (lane < 16) ? 0 : 8;
    const int arow  = blockM * 128 + am;

    // --- B staging: thread t loads W rows (2rp, 2rp+1) of each kstep,
    // cols cb..cb+7; packs bf16 pairs (K=2v,2v+1) and scatters dwords.
    // ISA 16-bit B 32x16 layout: lane<16 -> K {0..7,16..23}, lane>=16 -> K {8..15,24..31}.
    const int rp      = t >> 4;        // 0..15
    const int cb      = (t & 15) * 8;  // 0..120
    const int laneAdd = ((rp >> 2) & 1) * 16;
    const int vidx    = (rp & 3) + ((rp >> 3) << 2);

    v8f acc[4][2];
#pragma unroll
    for (int i = 0; i < 4; ++i)
#pragma unroll
        for (int j = 0; j < 2; ++j)
            acc[i][j] = v8f{0.f, 0.f, 0.f, 0.f, 0.f, 0.f, 0.f, 0.f};

#pragma unroll 1
    for (int kt = 0; kt < 32; ++kt) {
        // Dual-K: first 1024 over (A1,W1), second 1024 over (A2,W2). Uniform branch.
        const float* Ap;
        const float* Wp;
        int kb;
        if (kt < 16) { Ap = A1; Wp = W1; kb = kt * 64; }
        else         { Ap = A2; Wp = W2; kb = (kt - 16) * 64; }

        // ---- stage A slab (128x64 f32 -> bf16, fragment-swizzled) ----
        const float* asrc = Ap + (size_t)arow * DU + kb + ab0;
#pragma unroll
        for (int ks = 0; ks < 2; ++ks) {
            const float* s = asrc + ks * 32;
            f4 p0 = *(const f4*)(s);
            f4 p1 = *(const f4*)(s + 4);
            f4 p2 = *(const f4*)(s + 16);
            f4 p3 = *(const f4*)(s + 20);
            __bf16* adst = &ldsA[(ks * 256 + t) * 16];
#pragma unroll
            for (int e = 0; e < 4; ++e) {
                adst[e]      = (__bf16)p0[e];
                adst[4 + e]  = (__bf16)p1[e];
                adst[8 + e]  = (__bf16)p2[e];
                adst[12 + e] = (__bf16)p3[e];
            }
        }

        // ---- stage B slab (64x128 f32 -> bf16, transposed into fragments) ----
        unsigned* bdst = (unsigned*)ldsB;
#pragma unroll
        for (int ks = 0; ks < 2; ++ks) {
            const float* w0 = Wp + (size_t)(kb + ks * 32 + 2 * rp) * DU + blockN * 128 + cb;
            const float* w1 = w0 + DU;
            f4 q0 = *(const f4*)(w0);
            f4 q1 = *(const f4*)(w0 + 4);
            f4 q2 = *(const f4*)(w1);
            f4 q3 = *(const f4*)(w1 + 4);
#pragma unroll
            for (int j = 0; j < 8; ++j) {
                const int c  = cb + j;
                const int g  = c >> 4;
                const int ln = (c & 15) + laneAdd;
                const float lo = (j < 4) ? q0[j] : q1[j - 4];
                const float hi = (j < 4) ? q2[j] : q3[j - 4];
                bdst[((ks * 8 + g) * 32 + ln) * 8 + vidx] = pack_bf16x2(lo, hi);
            }
        }

        // Hint next slab into cache (uniform guard; no lane divergence).
        if ((kt & 15) != 15) {
            __builtin_prefetch(asrc + 64, 0, 0);
            __builtin_prefetch(Wp + (size_t)(kb + 64 + 2 * rp) * DU + blockN * 128 + cb, 0, 0);
        }

        __syncthreads();

        // ---- 2 K-steps x (4x2 frags) = 16 WMMAs per wave per stage ----
#pragma unroll
        for (int ks = 0; ks < 2; ++ks) {
            v16bf afr[4], bfr[2];
#pragma unroll
            for (int i = 0; i < 4; ++i)
                afr[i] = *(const v16bf*)&ldsA[((ks * 8 + waveM * 4 + i) * 32 + lane) * 16];
#pragma unroll
            for (int j = 0; j < 2; ++j)
                bfr[j] = *(const v16bf*)&ldsB[((ks * 8 + waveN * 2 + j) * 32 + lane) * 16];

#pragma unroll
            for (int i = 0; i < 4; ++i)
#pragma unroll
                for (int j = 0; j < 2; ++j)
                    acc[i][j] = __builtin_amdgcn_wmma_f32_16x16x32_bf16(
                        /*neg_a=*/false, afr[i], /*neg_b=*/false, bfr[j],
                        /*c_mod=*/(short)0, acc[i][j],
                        /*reuse_a=*/false, /*reuse_b=*/false);
        }

        __syncthreads();
    }

    // ---- epilogue (C frag layout: row = v + (lane<16?0:8), col = lane&15) ----
    const int colLocal = lane & 15;
    const int rowAdd   = (lane < 16) ? 0 : 8;
#pragma unroll
    for (int i = 0; i < 4; ++i) {
        const int rbase = blockM * 128 + waveM * 64 + i * 16 + rowAdd;
#pragma unroll
        for (int j = 0; j < 2; ++j) {
            const int col = blockN * 128 + waveN * 32 + j * 16 + colLocal;
            const float bv = bias[col];
#pragma unroll
            for (int v = 0; v < 8; ++v) {
                const size_t idx = (size_t)(rbase + v) * DU + col;
                const float pre = acc[i][j][v] + bv;
                if (MODE == 0) {
                    out[idx] = 1.0f / (1.0f + __expf(-pre));           // z
                } else if (MODE == 1) {
                    const float r = 1.0f / (1.0f + __expf(-pre));
                    out[idx] = r * hbuf[idx];                          // rh = r*h
                } else {
                    const float hc = tanhf(pre);
                    const float zv = zbuf[idx];
                    const float hv = hbuf[idx];
                    out[idx] = zv * hv + (1.0f - zv) * hc;             // h_next
                }
            }
        }
    }
}

extern "C" void kernel_launch(void* const* d_in, const int* in_sizes, int n_in,
                              void* d_out, int out_size, void* d_ws, size_t ws_size,
                              hipStream_t stream) {
    (void)in_sizes; (void)n_in; (void)out_size; (void)ws_size;

    const float* x     = (const float*)d_in[0];
    const float* h     = (const float*)d_in[1];
    const float* Wz_x  = (const float*)d_in[2];
    const float* Wz_h  = (const float*)d_in[3];
    const float* bz    = (const float*)d_in[4];
    const float* Wr_x  = (const float*)d_in[5];
    const float* Wr_h  = (const float*)d_in[6];
    const float* br    = (const float*)d_in[7];
    const float* Wh_x  = (const float*)d_in[8];
    const float* Wh_rh = (const float*)d_in[9];
    const float* bh    = (const float*)d_in[10];

    float* out = (float*)d_out;
    const size_t BU = (size_t)NROWS * DU;

    // scratch: z (16 MB) then rh (16 MB)
    float* zbuf  = (float*)d_ws;
    float* rhbuf = zbuf + BU;

    // first tuple element = old h
    hipMemcpyAsync(out, h, BU * sizeof(float), hipMemcpyDeviceToDevice, stream);

    dim3 grid(DU / 128, NROWS / 128); // 8 x 32
    dim3 block(256);

    gru_gemm<0><<<grid, block, 0, stream>>>(x, Wz_x, h, Wz_h, bz, nullptr, nullptr, zbuf);
    gru_gemm<1><<<grid, block, 0, stream>>>(x, Wr_x, h, Wr_h, br, h, nullptr, rhbuf);
    gru_gemm<2><<<grid, block, 0, stream>>>(x, Wh_x, rhbuf, Wh_rh, bh, h, zbuf, out + BU);
}
